// PoSGRU_17617955848420
// MI455X (gfx1250) — matrix operand order repaired and checked by reading, not compiled, not run on hardware
//
#include <hip/hip_runtime.h>
#include <hip/hip_bf16.h>
#include <math.h>

// ---------------------------------------------------------------------------
// PoSGRU for MI455X (gfx1250): bf16 WMMA GEMMs + batch-parallel GRU scan with
// async global->LDS double-buffered gate inputs.
// B=64, S=512, EMBED=HID=256, H=128, 3H=384, LAYERS=2, OUT=50.
// ---------------------------------------------------------------------------

typedef __attribute__((ext_vector_type(16))) __bf16 v16bf;
typedef __attribute__((ext_vector_type(8)))  float  v8f;
typedef int v4i __attribute__((vector_size(16)));   // matches builtin param type

#define BQ   64
#define SQ   512
#define HIDQ 256
#define HQ   128
#define G3Q  384
#define OUTQ 50
#define MRQ  (BQ * SQ)   // 32768 rows

#if defined(__has_builtin)
#  if __has_builtin(__builtin_amdgcn_global_load_async_to_lds_b128)
#    define HAVE_ASYNC_LDS 1
#  endif
#endif
#ifndef HAVE_ASYNC_LDS
#  define HAVE_ASYNC_LDS 0
#endif

// ---- WMMA fragment helpers (16x16x32 bf16, wave32) ------------------------
// A (16x32, 16-bit): lane<16 -> row=lane&15, elems 0..7 = K0..7, 8..15 =
// K16..23; lane>=16 -> +8 on both K bases (ISA 7.12.2 table).
// B (32x16, 16-bit): col = lane&15; same K order as A: elems 0..7 = K0..7,
// 8..15 = K16..23; lanes 16-31 -> +8 on both K bases.
// C/D: lanes 0-15 VGPR v -> M=v, lanes 16-31 -> M=v+8, N=lane&15.

__device__ __forceinline__ v16bf frag_a_from_f32(const float* A, int lda,
                                                 int row0, int k0, int lane) {
  const float* p = A + (size_t)(row0 + (lane & 15)) * lda + k0 + ((lane & 16) ? 8 : 0);
  v16bf a;
#pragma unroll
  for (int j = 0; j < 8; ++j) { a[j] = (__bf16)p[j]; a[j + 8] = (__bf16)p[16 + j]; }
  return a;
}

__device__ __forceinline__ v16bf frag_b_from_bf16(const __bf16* W, int ldw,
                                                  int n0, int k0, int lane) {
  const __bf16* p = W + (size_t)(n0 + (lane & 15)) * ldw + k0 + ((lane & 16) ? 8 : 0);
  v16bf b;
#pragma unroll
  for (int j = 0; j < 8; ++j) { b[j] = p[j]; b[j + 8] = p[16 + j]; }
  return b;
}

__device__ __forceinline__ v16bf frag_a_from_lds(const __bf16* sh, int stride,
                                                 int k0, int lane) {
  const __bf16* p = sh + (lane & 15) * stride + k0 + ((lane & 16) ? 8 : 0);
  v16bf a;
#pragma unroll
  for (int j = 0; j < 8; ++j) { a[j] = p[j]; a[j + 8] = p[16 + j]; }
  return a;
}

__device__ __forceinline__ v8f wmma_bf16(v16bf a, v16bf b, v8f c) {
  return __builtin_amdgcn_wmma_f32_16x16x32_bf16(false, a, false, b, (short)0, c,
                                                 false, false);
}

// ---- async wait helper -----------------------------------------------------
template <int N>
__device__ __forceinline__ void wait_async_le() {
#if HAVE_ASYNC_LDS
#  if __has_builtin(__builtin_amdgcn_s_wait_asynccnt)
  __builtin_amdgcn_s_wait_asynccnt((short)N);
#  else
  asm volatile("s_wait_asynccnt %0" ::"n"(N) : "memory");
#  endif
#endif
}

// ---- fp32 -> bf16 weight conversion ---------------------------------------
__global__ void f2b_kernel(const float* __restrict__ src, __bf16* __restrict__ dst,
                           int n) {
  int i = blockIdx.x * blockDim.x + threadIdx.x;
  if (i < n) dst[i] = (__bf16)src[i];
}

// ---- embedding gather (padding row already zero in table) -----------------
__global__ void embed_kernel(const int* __restrict__ x, const float* __restrict__ emb,
                             float* __restrict__ out, int rows) {
  int i = blockIdx.x * blockDim.x + threadIdx.x;   // over rows*256 elements
  if (i < rows * HIDQ) {
    int r = i >> 8, c = i & 255;
    out[i] = emb[(size_t)x[r] * HIDQ + c];
  }
}

// ---- generic WMMA GEMM: C[M,N] = A[M,256] @ W[N,256]^T + bias -------------
// act: 0 = none, 1 = exact-erf GELU.  outF (fp32) and/or outB (bf16).
__global__ __launch_bounds__(256) void gemm_kernel(
    const float* __restrict__ A, const __bf16* __restrict__ W,
    const float* __restrict__ bias, int M, int N,
    float* __restrict__ outF, __bf16* __restrict__ outB, int act) {
  const int lane = threadIdx.x & 31;
  const int wv   = threadIdx.x >> 5;
  const int m0   = blockIdx.x * 128 + wv * 16;   // 8 waves x 16 rows
  if (m0 >= M) return;                            // wave-uniform

  // Hold the full 16x256 bf16 A strip in VGPRs (8 K-fragments).
  v16bf Af[8];
#pragma unroll
  for (int kt = 0; kt < 8; ++kt)
    Af[kt] = frag_a_from_f32(A, HIDQ, m0, kt * 32, lane);

  const int Npad = (N + 15) & ~15;
  for (int n0 = 0; n0 < Npad; n0 += 16) {
    v8f c = {};
#pragma unroll
    for (int kt = 0; kt < 8; ++kt) {
      v16bf b = frag_b_from_bf16(W, HIDQ, n0, kt * 32, lane);
      c = wmma_bf16(Af[kt], b, c);
    }
    const int n  = n0 + (lane & 15);
    const int mb = m0 + ((lane & 16) ? 8 : 0);
    const float bv = (bias && n < N) ? bias[n] : 0.0f;
#pragma unroll
    for (int v = 0; v < 8; ++v) {
      float xv = c[v] + bv;
      if (act == 1) xv = 0.5f * xv * (1.0f + erff(xv * 0.70710678118654752f));
      if (n < N) {
        size_t o = (size_t)(mb + v) * N + n;
        if (outF) outF[o] = xv;
        if (outB) outB[o] = (__bf16)xv;
      }
    }
  }
}

// ---- bidirectional GRU scan -----------------------------------------------
// 8 blocks x 256 threads: blockIdx.x = dir*4 + bgroup.  Each block owns 16
// batch rows of one direction.  Per step: hp = h @ Whh^T via WMMA (Whh
// fragments pinned in VGPRs for all 512 steps), PyTorch (r,z,n) gates, output
// h_out = out + h_in residual.  Gate inputs xp[t] are double-buffered in LDS
// via GLOBAL_LOAD_ASYNC_TO_LDS_B128 issued one step ahead (ASYNCcnt pipelined).
__global__ __launch_bounds__(256) void gru_scan_kernel(
    const __bf16* __restrict__ xpF, const __bf16* __restrict__ xpB,
    const __bf16* __restrict__ WhhF, const __bf16* __restrict__ WhhB,
    const float* __restrict__ bhhF, const float* __restrict__ bhhB,
    const float* __restrict__ h_in, float* __restrict__ h_out) {
  const int dir = blockIdx.x >> 2;
  const int bg  = blockIdx.x & 3;
  const __bf16* xp  = dir ? xpB  : xpF;
  const __bf16* Whh = dir ? WhhB : WhhF;
  const float*  bhh = dir ? bhhB : bhhF;

  __shared__ float  sh_hf[16][132];      // fp32 hidden state
  __shared__ __bf16 sh_hb[16][136];      // bf16 copy for WMMA A-fragments
  __shared__ float  sh_hp[16][388];      // hp = h @ Whh^T (384 gate cols)
  __shared__ __bf16 sh_xp[2][16][392];   // double-buffered xp[t] tiles

  const int lane = threadIdx.x & 31;
  const int wv   = threadIdx.x >> 5;
  const int n0w  = wv * 48;              // this wave's 48 gate columns

  // Pin Whh fragments (3 N-tiles x 4 K-tiles = 96 VGPRs) for the whole scan.
  v16bf Bf[3][4];
#pragma unroll
  for (int nt = 0; nt < 3; ++nt)
#pragma unroll
    for (int kt = 0; kt < 4; ++kt)
      Bf[nt][kt] = frag_b_from_bf16(Whh, HQ, n0w + nt * 16, kt * 32, lane);

  // h0 = 0
  for (int i = threadIdx.x; i < 16 * 132; i += 256) (&sh_hf[0][0])[i] = 0.0f;
  for (int i = threadIdx.x; i < 16 * 136; i += 256) (&sh_hb[0][0])[i] = (__bf16)0.0f;

  // Per-thread gate bias registers (thread owns 8 fixed (b,h) cells).
  float cbr[8], cbz[8], cbn[8];
#pragma unroll
  for (int e = 0; e < 8; ++e) {
    int hh = (threadIdx.x * 8 + e) & 127;
    cbr[e] = bhh[hh]; cbz[e] = bhh[128 + hh]; cbn[e] = bhh[256 + hh];
  }

  // Precompute this thread's 3 copy chunks (16 rows x 768B = 768 x 16B).
  int srow[3], scol[3];
  const __bf16* sgp[3];
#pragma unroll
  for (int it = 0; it < 3; ++it) {
    int i = threadIdx.x + it * 256;
    srow[it] = i / 48;
    scol[it] = (i % 48) * 8;
    sgp[it]  = xp + (size_t)(bg * 16 + srow[it]) * SQ * G3Q + scol[it];
  }

  // Stage helper: copy xp tile for timestep t into sh_xp[buf].
  auto stage_xp = [&](int t, int buf) {
#pragma unroll
    for (int it = 0; it < 3; ++it) {
      const __bf16* g = sgp[it] + (size_t)t * G3Q;
      __bf16* l = &sh_xp[buf][srow[it]][scol[it]];
#if HAVE_ASYNC_LDS
      __builtin_amdgcn_global_load_async_to_lds_b128(
          (__attribute__((address_space(1))) v4i*)const_cast<__bf16*>(g),
          (__attribute__((address_space(3))) v4i*)l, 0, 0);
#else
      ((uint4*)l)[0] = ((const uint4*)g)[0];
#endif
    }
  };

  // Prologue: stage t0 into buffer 0 (waited via asynccnt<=3 in step 0).
  stage_xp(dir ? (SQ - 1) : 0, 0);
  __syncthreads();

  for (int ti = 0; ti < SQ; ++ti) {
    const int t   = dir ? (SQ - 1 - ti) : ti;
    const int cur = ti & 1;

    // Issue next step's xp copy immediately; it overlaps WMMA + gate work.
    if (ti + 1 < SQ) stage_xp(dir ? (t - 1) : (t + 1), cur ^ 1);

    // hp = h_state @ Whh^T  (12 WMMAs per wave)
    v16bf Af[4];
#pragma unroll
    for (int kt = 0; kt < 4; ++kt)
      Af[kt] = frag_a_from_lds(&sh_hb[0][0], 136, kt * 32, lane);
#pragma unroll
    for (int nt = 0; nt < 3; ++nt) {
      v8f c = {};
#pragma unroll
      for (int kt = 0; kt < 4; ++kt) c = wmma_bf16(Af[kt], Bf[nt][kt], c);
      const int ncol = n0w + nt * 16 + (lane & 15);
      const int mrow = (lane & 16) ? 8 : 0;
#pragma unroll
      for (int v = 0; v < 8; ++v) sh_hp[mrow + v][ncol] = c[v];
    }

    // Older async group (this step's xp) must have landed: loads complete in
    // order, so asynccnt<=3 leaves only the newest 3 (next step's) in flight.
    if (ti + 1 < SQ) wait_async_le<3>();
    else             wait_async_le<0>();
    __syncthreads();

    // Gate math: 16x128 cells, 8 consecutive per thread.
#pragma unroll
    for (int e = 0; e < 8; ++e) {
      const int idx = threadIdx.x * 8 + e;
      const int b = idx >> 7, hh = idx & 127;
      const float xg_r = (float)sh_xp[cur][b][hh];
      const float xg_z = (float)sh_xp[cur][b][128 + hh];
      const float xg_n = (float)sh_xp[cur][b][256 + hh];
      const float hg_r = sh_hp[b][hh]       + cbr[e];
      const float hg_z = sh_hp[b][128 + hh] + cbz[e];
      const float hg_n = sh_hp[b][256 + hh] + cbn[e];
      const float r  = 1.0f / (1.0f + expf(-(xg_r + hg_r)));
      const float z  = 1.0f / (1.0f + expf(-(xg_z + hg_z)));
      const float nn = tanhf(xg_n + r * hg_n);
      const float hprev = sh_hf[b][hh];
      const float hnew  = (1.0f - z) * nn + z * hprev;
      sh_hf[b][hh] = hnew;
      sh_hb[b][hh] = (__bf16)hnew;
      const size_t o = ((size_t)(bg * 16 + b) * SQ + t) * HIDQ + dir * HQ + hh;
      h_out[o] = hnew + h_in[o];                   // residual half
    }
    __syncthreads();
  }
}

// ---------------------------------------------------------------------------
extern "C" void kernel_launch(void* const* d_in, const int* in_sizes, int n_in,
                              void* d_out, int out_size, void* d_ws, size_t ws_size,
                              hipStream_t stream) {
  const int*   x    = (const int*)d_in[0];
  const float* emb  = (const float*)d_in[1];
  const float* W1   = (const float*)d_in[2];
  const float* b1   = (const float*)d_in[3];
  const float* Wih  = (const float*)d_in[4];   // [2,2,384,256]
  const float* Whh  = (const float*)d_in[5];   // [2,2,384,128]
  const float* bih  = (const float*)d_in[6];   // [2,2,384]
  const float* bhh  = (const float*)d_in[7];   // [2,2,384]
  const float* W2   = (const float*)d_in[8];
  const float* b2   = (const float*)d_in[9];
  const float* W3   = (const float*)d_in[10];  // [50,256]
  const float* b3   = (const float*)d_in[11];

  // Workspace layout (~119 MB)
  char* ws = (char*)d_ws;
  const size_t szH  = (size_t)MRQ * HIDQ * sizeof(float);   // 33.5 MB
  const size_t szXP = (size_t)MRQ * G3Q * sizeof(__bf16);   // 25.2 MB
  float*  hA   = (float*)ws;
  float*  hB   = (float*)(ws + szH);
  __bf16* xpF  = (__bf16*)(ws + 2 * szH);
  __bf16* xpB  = (__bf16*)(ws + 2 * szH + szXP);
  __bf16* pool = (__bf16*)(ws + 2 * szH + 2 * szXP);
  __bf16* W1b  = pool;                 // 65536
  __bf16* Wihb = W1b  + 65536;         // 4 x 98304
  __bf16* Whhb = Wihb + 393216;        // 4 x 49152
  __bf16* W2b  = Whhb + 196608;        // 65536
  __bf16* W3b  = W2b  + 65536;         // 64x256 padded (rows 50..63 unused)

  // Weight conversions (fp32 -> bf16)
  f2b_kernel<<<(65536  + 255) / 256, 256, 0, stream>>>(W1,  W1b,  65536);
  f2b_kernel<<<(393216 + 255) / 256, 256, 0, stream>>>(Wih, Wihb, 393216);
  f2b_kernel<<<(196608 + 255) / 256, 256, 0, stream>>>(Whh, Whhb, 196608);
  f2b_kernel<<<(65536  + 255) / 256, 256, 0, stream>>>(W2,  W2b,  65536);
  f2b_kernel<<<(12800  + 255) / 256, 256, 0, stream>>>(W3,  W3b,  12800);

  // h = embed[x] @ W1^T + b1
  embed_kernel<<<(MRQ * HIDQ + 255) / 256, 256, 0, stream>>>(x, emb, hB, MRQ);
  gemm_kernel<<<MRQ / 128, 256, 0, stream>>>(hB, W1b, b1, MRQ, HIDQ, hA, nullptr, 0);

  float* hcur = hA;
  float* hnxt = hB;
  for (int l = 0; l < 2; ++l) {
    // Hoisted input projections for both directions (bf16 xp buffers).
    gemm_kernel<<<MRQ / 128, 256, 0, stream>>>(
        hcur, Wihb + (size_t)(l * 2 + 0) * 98304, bih + (l * 2 + 0) * 384,
        MRQ, G3Q, nullptr, xpF, 0);
    gemm_kernel<<<MRQ / 128, 256, 0, stream>>>(
        hcur, Wihb + (size_t)(l * 2 + 1) * 98304, bih + (l * 2 + 1) * 384,
        MRQ, G3Q, nullptr, xpB, 0);
    // Both directions scanned concurrently; writes disjoint halves + residual.
    gru_scan_kernel<<<8, 256, 0, stream>>>(
        xpF, xpB,
        Whhb + (size_t)(l * 2 + 0) * 49152, Whhb + (size_t)(l * 2 + 1) * 49152,
        bhh + (l * 2 + 0) * 384, bhh + (l * 2 + 1) * 384, hcur, hnxt);
    float* tsw = hcur; hcur = hnxt; hnxt = tsw;
  }

  // GELU MLP head: gelu(h@W2^T+b2) @ W3^T + b3  -> d_out [B*S,50]
  gemm_kernel<<<MRQ / 128, 256, 0, stream>>>(hcur, W2b, b2, MRQ, HIDQ, hnxt, nullptr, 1);
  gemm_kernel<<<MRQ / 128, 256, 0, stream>>>(hnxt, W3b, b3, MRQ, OUTQ,
                                             (float*)d_out, nullptr, 0);
}
